// PennyLaneHead_23055384445012
// MI455X (gfx1250) — matrix operand order repaired and checked by reading, not compiled, not run on hardware
//
#include <hip/hip_runtime.h>
#include <hip/hip_bf16.h>
#include <math.h>

typedef __attribute__((ext_vector_type(16))) _Float16 v16h;
typedef __attribute__((ext_vector_type(8)))  float    v8f;

#define BATCH   4096
#define FEAT    768
#define NQ      12
#define NCLASS  10
#define DEPTH   3
#define DIM     4096   // 2^NQ
#define HALF_PI 1.57079632679489662f

// ---------------------------------------------------------------------------
// Kernel 1: angles = tanh(x @ Wp^T) * pi/2, via v_wmma_f32_16x16x32_f16.
// Grid: 32 blocks x 256 threads (8 waves/block); each wave owns one 16-row
// output tile (256 tiles total). N=12 padded to one 16-wide WMMA tile.
// Padded lanes use a 0/1 multiplicative mask (no divergent branch in K loop).
// ---------------------------------------------------------------------------
__global__ __launch_bounds__(256)
void proj_tanh_wmma(const float* __restrict__ x,
                    const float* __restrict__ Wp,
                    float* __restrict__ angles)
{
    const int wave = threadIdx.x >> 5;
    const int lane = threadIdx.x & 31;
    const int tile = blockIdx.x * 8 + wave;     // 0..255
    const int row_base = tile * 16;

    const int half = lane >> 4;                 // 0: lanes 0-15, 1: lanes 16-31
    const int mrow = lane & 15;                 // A-matrix row for this lane
    const int ncol = lane & 15;                 // B/C/D column for this lane

    v8f acc = {};

    const float* xrow = x + (size_t)(row_base + mrow) * FEAT;
    const int   wrowi = (ncol < NQ) ? ncol : (NQ - 1);       // safe row
    const float nmask = (ncol < NQ) ? 1.0f : 0.0f;           // zero pad lanes
    const float* wrow = Wp + (size_t)wrowi * FEAT;

    #pragma unroll 2
    for (int k0 = 0; k0 < FEAT; k0 += 32) {
        // prefetch next A tile (emits global_prefetch_b8)
        if (k0 + 32 < FEAT) __builtin_prefetch(xrow + k0 + 32, 0, 0);

        // ---- A tile (16x32 f16). Per ISA layout:
        // half 0: elems 0-7 -> K=0..7,  elems 8-15 -> K=16..23
        // half 1: elems 0-7 -> K=8..15, elems 8-15 -> K=24..31
        const float4 a0 = *(const float4*)(xrow + k0 + 8 * half);
        const float4 a1 = *(const float4*)(xrow + k0 + 8 * half + 4);
        const float4 a2 = *(const float4*)(xrow + k0 + 16 + 8 * half);
        const float4 a3 = *(const float4*)(xrow + k0 + 16 + 8 * half + 4);
        v16h a;
        a[0]=(_Float16)a0.x; a[1]=(_Float16)a0.y; a[2]=(_Float16)a0.z; a[3]=(_Float16)a0.w;
        a[4]=(_Float16)a1.x; a[5]=(_Float16)a1.y; a[6]=(_Float16)a1.z; a[7]=(_Float16)a1.w;
        a[8]=(_Float16)a2.x; a[9]=(_Float16)a2.y; a[10]=(_Float16)a2.z; a[11]=(_Float16)a2.w;
        a[12]=(_Float16)a3.x; a[13]=(_Float16)a3.y; a[14]=(_Float16)a3.z; a[15]=(_Float16)a3.w;

        // ---- B tile (32x16 f16), B[k][n] = Wp[n][k]. Lane = column n;
        // element e -> K = 8*half + e (e < 8), K = 16 + 8*half + (e - 8) (e >= 8).
        const float4 b0 = *(const float4*)(wrow + k0 + 8 * half);
        const float4 b1 = *(const float4*)(wrow + k0 + 8 * half + 4);
        const float4 b2 = *(const float4*)(wrow + k0 + 16 + 8 * half);
        const float4 b3 = *(const float4*)(wrow + k0 + 16 + 8 * half + 4);
        v16h b;
        b[0]=(_Float16)(b0.x*nmask); b[1]=(_Float16)(b0.y*nmask);
        b[2]=(_Float16)(b0.z*nmask); b[3]=(_Float16)(b0.w*nmask);
        b[4]=(_Float16)(b1.x*nmask); b[5]=(_Float16)(b1.y*nmask);
        b[6]=(_Float16)(b1.z*nmask); b[7]=(_Float16)(b1.w*nmask);
        b[8]=(_Float16)(b2.x*nmask); b[9]=(_Float16)(b2.y*nmask);
        b[10]=(_Float16)(b2.z*nmask); b[11]=(_Float16)(b2.w*nmask);
        b[12]=(_Float16)(b3.x*nmask); b[13]=(_Float16)(b3.y*nmask);
        b[14]=(_Float16)(b3.z*nmask); b[15]=(_Float16)(b3.w*nmask);

        acc = __builtin_amdgcn_wmma_f32_16x16x32_f16(
                  false, a, false, b, (short)0, acc, false, false);
    }

    // C/D layout: lane -> column n; VGPR v -> row (v + 8*half).
    if (ncol < NQ) {
        #pragma unroll
        for (int v = 0; v < 8; v++) {
            int m = row_base + v + 8 * half;
            angles[m * NQ + ncol] = tanhf(acc[v]) * HALF_PI;
        }
    }
}

// ---------------------------------------------------------------------------
// Kernel 2: full 12-qubit circuit per sample, state resident in LDS (32 KB).
// One workgroup (256 threads, 8 waves) per batch sample.
// Fuses: product-state RY init, 3x(12 Rot + fused ring-CNOT permutation),
// PauliZ expvals, and the final [12]->[10] linear head.
// Each CNOT ring layer is a GF(2)-linear index map applied as ONE gather pass
// (registers carry the permuted state across a single barrier).
// ---------------------------------------------------------------------------
__global__ __launch_bounds__(256)
void circuit_kernel(const float* __restrict__ angles,
                    const float* __restrict__ w,      // [DEPTH][NQ][3]
                    const float* __restrict__ Wo,     // [NCLASS][NQ]
                    const float* __restrict__ bo,     // [NCLASS]
                    float* __restrict__ out)          // [BATCH][NCLASS]
{
    __shared__ float sre[DIM];
    __shared__ float sim[DIM];
    __shared__ float gm[DEPTH][NQ][8];   // complex 2x2 per Rot gate
    __shared__ float enc_c[NQ], enc_s[NQ];
    __shared__ float zred[8 * 16];       // per-wave partial PauliZ sums
    __shared__ float zfin[NQ];

    const int tid = threadIdx.x;
    const int smp = blockIdx.x;

    // Per-sample encoding angles -> cos/sin half-angles.
    if (tid < NQ) {
        float a = angles[smp * NQ + tid] * 0.5f;
        enc_c[tid] = cosf(a);
        enc_s[tid] = sinf(a);
    }
    // Rot(phi,theta,omega) = RZ(omega) RY(theta) RZ(phi) matrices (36 gates).
    if (tid >= 32 && tid < 32 + DEPTH * NQ) {
        int g = tid - 32, l = g / NQ, q = g % NQ;
        const float* wg = w + (l * NQ + q) * 3;
        float phi = wg[0], th = wg[1], om = wg[2];
        float ct = cosf(0.5f * th), st = sinf(0.5f * th);
        float ap = 0.5f * (phi + om), am = 0.5f * (phi - om);
        float cap = cosf(ap), sap = sinf(ap);
        float cam = cosf(am), sam = sinf(am);
        gm[l][q][0] =  cap * ct;  gm[l][q][1] = -sap * ct;   // m00 = e^{-i ap} c
        gm[l][q][2] = -cam * st;  gm[l][q][3] = -sam * st;   // m01 = -e^{+i am} s
        gm[l][q][4] =  cam * st;  gm[l][q][5] = -sam * st;   // m10 = e^{-i am} s
        gm[l][q][6] =  cap * ct;  gm[l][q][7] =  sap * ct;   // m11 = e^{+i ap} c
    }
    __syncthreads();

    // Product-state init: amp[k] = prod_q (bit_q(k) ? s_q : c_q); wire0 = MSB.
    #pragma unroll 4
    for (int j = 0; j < DIM / 256; j++) {
        int idx = tid + 256 * j;
        float amp = 1.0f;
        #pragma unroll
        for (int q = 0; q < NQ; q++)
            amp *= ((idx >> (NQ - 1 - q)) & 1) ? enc_s[q] : enc_c[q];
        sre[idx] = amp;
        sim[idx] = 0.0f;
    }
    __syncthreads();

    for (int l = 0; l < DEPTH; l++) {
        // ---- 12 Rot gates (2x2 complex on wire q; bit position 11-q) ----
        for (int q = 0; q < NQ; q++) {
            const int bpos = NQ - 1 - q;
            const int mask = 1 << bpos;
            const float m00r = gm[l][q][0], m00i = gm[l][q][1];
            const float m01r = gm[l][q][2], m01i = gm[l][q][3];
            const float m10r = gm[l][q][4], m10i = gm[l][q][5];
            const float m11r = gm[l][q][6], m11i = gm[l][q][7];
            #pragma unroll
            for (int j = 0; j < (DIM / 2) / 256; j++) {
                int p = tid + 256 * j;                       // pair index
                int lo = p & (mask - 1);
                int i0 = ((p ^ lo) << 1) | lo;               // bit bpos = 0
                int i1 = i0 | mask;                          // bit bpos = 1
                float r0 = sre[i0], ii0 = sim[i0];
                float r1 = sre[i1], ii1 = sim[i1];
                sre[i0] = m00r*r0 - m00i*ii0 + m01r*r1 - m01i*ii1;
                sim[i0] = m00r*ii0 + m00i*r0 + m01r*ii1 + m01i*r1;
                sre[i1] = m10r*r0 - m10i*ii0 + m11r*r1 - m11i*ii1;
                sim[i1] = m10r*ii0 + m10i*r0 + m11r*ii1 + m11i*r1;
            }
            __syncthreads();
        }

        // ---- fused ring-CNOT layer as one GF(2) permutation gather ----
        // new[y] = old[N y] with N = F_1 * F_2 * ... * F_12 built by
        // col[cb] ^= col[tb] in gate order (F = flip tgt bit if ctrl set).
        const int r = l % (NQ - 1) + 1;
        int col[NQ];
        #pragma unroll
        for (int j = 0; j < NQ; j++) col[j] = 1 << j;
        #pragma unroll
        for (int q = 0; q < NQ; q++) {
            int t  = (q + r) % NQ;
            int cb = NQ - 1 - q, tb = NQ - 1 - t;
            col[cb] ^= col[tb];
        }
        // source index for y = tid + 256*j : x = xor of col[bit] over set bits
        int xbase = 0;
        #pragma unroll
        for (int bpos = 0; bpos < 8; bpos++)
            if ((tid >> bpos) & 1) xbase ^= col[bpos];
        float tr[DIM / 256], ti[DIM / 256];
        #pragma unroll
        for (int j = 0; j < DIM / 256; j++) {
            int xs = xbase;
            if (j & 1) xs ^= col[8];
            if (j & 2) xs ^= col[9];
            if (j & 4) xs ^= col[10];
            if (j & 8) xs ^= col[11];
            tr[j] = sre[xs];
            ti[j] = sim[xs];
        }
        __syncthreads();
        #pragma unroll
        for (int j = 0; j < DIM / 256; j++) {
            int y = tid + 256 * j;
            sre[y] = tr[j];
            sim[y] = ti[j];
        }
        __syncthreads();
    }

    // ---- PauliZ expectations: z_q = sum_k sign_q(k) * |amp_k|^2 ----
    float zp[NQ];
    #pragma unroll
    for (int q = 0; q < NQ; q++) zp[q] = 0.0f;
    #pragma unroll 4
    for (int j = 0; j < DIM / 256; j++) {
        int idx = tid + 256 * j;
        float pr = sre[idx] * sre[idx] + sim[idx] * sim[idx];
        #pragma unroll
        for (int q = 0; q < NQ; q++)
            zp[q] += ((idx >> (NQ - 1 - q)) & 1) ? -pr : pr;
    }
    // wave32 reduction, then cross-wave via LDS
    #pragma unroll
    for (int off = 16; off >= 1; off >>= 1) {
        #pragma unroll
        for (int q = 0; q < NQ; q++)
            zp[q] += __shfl_down(zp[q], off, 32);
    }
    if ((tid & 31) == 0) {
        int wave = tid >> 5;
        #pragma unroll
        for (int q = 0; q < NQ; q++) zred[wave * 16 + q] = zp[q];
    }
    __syncthreads();
    if (tid < NQ) {
        float s = 0.0f;
        #pragma unroll
        for (int wv = 0; wv < 8; wv++) s += zred[wv * 16 + tid];
        zfin[tid] = s;
    }
    __syncthreads();

    // ---- fused output head: out = Wo @ z + bo ----
    if (tid < NCLASS) {
        float o = bo[tid];
        #pragma unroll
        for (int q = 0; q < NQ; q++) o += Wo[tid * NQ + q] * zfin[q];
        out[smp * NCLASS + tid] = o;
    }
}

extern "C" void kernel_launch(void* const* d_in, const int* in_sizes, int n_in,
                              void* d_out, int out_size, void* d_ws, size_t ws_size,
                              hipStream_t stream) {
    (void)in_sizes; (void)n_in; (void)out_size; (void)ws_size;
    const float* x       = (const float*)d_in[0];  // [4096, 768]
    const float* Wp      = (const float*)d_in[1];  // [12, 768]
    const float* weights = (const float*)d_in[2];  // [3, 12, 3]
    const float* Wo      = (const float*)d_in[3];  // [10, 12]
    const float* bo      = (const float*)d_in[4];  // [10]
    float* out    = (float*)d_out;                 // [4096, 10]
    float* angles = (float*)d_ws;                  // [4096, 12] scratch (192 KB)

    proj_tanh_wmma<<<BATCH / 16 / 8, 256, 0, stream>>>(x, Wp, angles);
    circuit_kernel<<<BATCH, 256, 0, stream>>>(angles, weights, Wo, bo, out);
}
